// SpLoRaLinear_61134564491363
// MI455X (gfx1250) — matrix operand order repaired and checked by reading, not compiled, not run on hardware
//
#include <hip/hip_runtime.h>

// ---------------- problem constants ----------------
#define IN_F   4096
#define OUT_F  4096
#define RANK   128
#define N_TOK  8192
#define KX     (IN_F + RANK)          // 4224, concatenated K
#define SCALE  (1.0f / 128.0f)

typedef __attribute__((ext_vector_type(16))) __bf16 v16bf;
typedef __attribute__((ext_vector_type(8)))  __bf16 v8bf;
typedef __attribute__((ext_vector_type(8)))  float  v8f;

// ---------------- init kernels ----------------

// Wcat[o][0..4095] = 0 ; Wcat[o][4096+r] = bf16(lora_B[o][r])
__global__ __launch_bounds__(256) void init_wcat(const float* __restrict__ B,
                                                 __bf16* __restrict__ Wcat) {
  long t = (long)blockIdx.x * blockDim.x + threadIdx.x;   // over OUT_F*KX/4
  long total = (long)OUT_F * (KX / 4);
  if (t >= total) return;
  long o  = t / (KX / 4);
  int  kc = (int)(t % (KX / 4)) * 4;
  union { __bf16 b[4]; unsigned long long u; } cv;
  cv.u = 0ULL;
  if (kc >= IN_F) {
    int r = kc - IN_F;
    #pragma unroll
    for (int j = 0; j < 4; ++j) cv.b[j] = (__bf16)B[o * RANK + r + j];
  }
  *(unsigned long long*)&Wcat[o * (long)KX + kc] = cv.u;
}

// Abf[r][k] = bf16(lora_A[r][k] * SCALE)
__global__ __launch_bounds__(256) void init_abf(const float* __restrict__ A,
                                                __bf16* __restrict__ Abf) {
  long t = (long)blockIdx.x * blockDim.x + threadIdx.x;   // over RANK*IN_F/4
  if (t >= (long)RANK * (IN_F / 4)) return;
  const float4 f = ((const float4*)A)[t];
  union { __bf16 b[4]; unsigned long long u; } cv;
  cv.b[0] = (__bf16)(f.x * SCALE); cv.b[1] = (__bf16)(f.y * SCALE);
  cv.b[2] = (__bf16)(f.z * SCALE); cv.b[3] = (__bf16)(f.w * SCALE);
  *(unsigned long long*)&Abf[t * 4] = cv.u;
}

// Xcat[n][0..4095] = bf16(x[n][k])  (cols 4096.. filled later by LoRA GEMM)
__global__ __launch_bounds__(256) void init_xcat(const float* __restrict__ x,
                                                 __bf16* __restrict__ Xcat) {
  long t = (long)blockIdx.x * blockDim.x + threadIdx.x;   // over N_TOK*IN_F/4
  if (t >= (long)N_TOK * (IN_F / 4)) return;
  long n  = t / (IN_F / 4);
  int  kc = (int)(t % (IN_F / 4)) * 4;
  const float4 f = ((const float4*)x)[t];
  union { __bf16 b[4]; unsigned long long u; } cv;
  cv.b[0] = (__bf16)f.x; cv.b[1] = (__bf16)f.y;
  cv.b[2] = (__bf16)f.z; cv.b[3] = (__bf16)f.w;
  *(unsigned long long*)&Xcat[n * (long)KX + kc] = cv.u;
}

// sorted-COO scatter: duplicates are adjacent -> run leader sums, plain store.
__global__ __launch_bounds__(256) void scatter_sorted(const float* __restrict__ vals,
                                                      const int* __restrict__ idx,
                                                      __bf16* __restrict__ Wcat,
                                                      int nnz) {
  int j = blockIdx.x * blockDim.x + threadIdx.x;
  if (j >= nnz) return;
  int me = idx[j];
  if (j > 0 && idx[j - 1] == me) return;      // not the run leader
  float s = vals[j];
  int k = j + 1;
  while (k < nnz && idx[k] == me) { s += vals[k]; ++k; }
  int o = me >> 12;         // / IN_F (4096)
  int i = me & (IN_F - 1);  // % IN_F
  Wcat[(long)o * KX + i] = (__bf16)s;
}

// async global -> LDS (GV mode), tracked by ASYNCcnt
__device__ __forceinline__ void async_copy_b128(unsigned lds_off, const void* gaddr) {
  asm volatile("global_load_async_to_lds_b128 %0, %1, off"
               :: "v"(lds_off), "v"(gaddr) : "memory");
}
__device__ __forceinline__ void wait_asynccnt0() {
  asm volatile("s_wait_asynccnt 0x0" ::: "memory");
}

// ---------------- WMMA GEMM ----------------
// C[M,N] = X[M,K] * W[N,K]^T   (both K-major / row-major with contiguous K)
// block tile 128x128, 8 waves in 2(M) x 4(N), wave tile 64x32.
// F32OUT: f32 store with bias; else bf16 store.
template <bool F32OUT>
__global__ __launch_bounds__(256)
void gemm_wmma_bf16(const __bf16* __restrict__ X, long ldx,
                    const __bf16* __restrict__ W, long ldw,
                    int kIters,
                    float* __restrict__ outF,
                    __bf16* __restrict__ outB, long ldo,
                    const float* __restrict__ bias) {
  __shared__ __align__(16) __bf16 sX[2][128][40];   // 40-elem stride: conflict-free
  __shared__ __align__(16) __bf16 sW[2][128][40];
  constexpr int BUF_ELEMS = 128 * 40;               // elems per buffer
  constexpr unsigned BUF_BYTES = BUF_ELEMS * 2;     // 10240

  const int tid  = threadIdx.x;
  const int lane = tid & 31;
  const int wave = tid >> 5;
  const int wr = wave >> 2;                 // 0..1  (M)
  const int wc = wave & 3;                  // 0..3  (N)
  const long blockM = (long)blockIdx.x * 128;
  const long blockN = (long)blockIdx.y * 128;

  // global->LDS assignment: row = tid/2, segment of 16 bf16 = (tid&1)*16
  const int lr  = tid >> 1;
  const int lsg = (tid & 1) * 16;
  const __bf16* gX = X + (blockM + lr) * ldx + lsg;
  const __bf16* gW = W + (blockN + lr) * ldw + lsg;
  const unsigned ldsX0 = (unsigned)(uintptr_t)&sX[0][lr][lsg];
  const unsigned ldsW0 = (unsigned)(uintptr_t)&sW[0][lr][lsg];

  // preload stage 0 via async DMA into LDS buffer 0
  async_copy_b128(ldsX0, gX);
  async_copy_b128(ldsW0, gW);
  wait_asynccnt0();
  __syncthreads();

  v8f acc[2][4] = {};

  // fragment lane geometry (per ISA 16-bit A 16x32 / B 32x16 layouts)
  const int mA = lane & 15;                  // A row within 16
  const int c1 = (lane < 16) ? 0 : 8;        // A: K chunk1 (chunk2 = +16)
  const int nB = lane & 15;                  // B column
  const int kB = (lane < 16) ? 0 : 8;        // B: K chunk1 (chunk2 = +16)

  const __bf16* pX0 = &sX[0][wr * 64 + mA][c1];   // + mt*16 rows
  const __bf16* pW0 = &sW[0][wc * 32 + nB][kB];   // + nt*16 rows

  for (int ks = 0; ks < kIters; ++ks) {
    const int buf = ks & 1;

    // kick off async fill of the other buffer (safe: barrier at end of
    // previous stage guarantees no wave still reads buf^1)
    if (ks + 1 < kIters) {
      const unsigned dst = (unsigned)(buf ^ 1) * BUF_BYTES;
      async_copy_b128(ldsX0 + dst, gX + (long)(ks + 1) * 32);
      async_copy_b128(ldsW0 + dst, gW + (long)(ks + 1) * 32);
    }

    // load all fragments first so the 8 WMMAs can issue back-to-back
    const __bf16* bx = pX0 + buf * BUF_ELEMS;
    const __bf16* bw = pW0 + buf * BUF_ELEMS;
    v16bf bfrag[2];
    #pragma unroll
    for (int nt = 0; nt < 2; ++nt) {
      const __bf16* p = bw + nt * 16 * 40;
      v8bf lo = *(const v8bf*)p;
      v8bf hi = *(const v8bf*)(p + 16);
      bfrag[nt] = __builtin_shufflevector(lo, hi, 0,1,2,3,4,5,6,7,8,9,10,11,12,13,14,15);
    }
    v16bf afrag[4];
    #pragma unroll
    for (int mt = 0; mt < 4; ++mt) {
      const __bf16* p = bx + mt * 16 * 40;
      v8bf lo = *(const v8bf*)p;
      v8bf hi = *(const v8bf*)(p + 16);
      afrag[mt] = __builtin_shufflevector(lo, hi, 0,1,2,3,4,5,6,7,8,9,10,11,12,13,14,15);
    }

    #pragma unroll
    for (int mt = 0; mt < 4; ++mt) {
      #pragma unroll
      for (int nt = 0; nt < 2; ++nt) {
        acc[nt][mt] = __builtin_amdgcn_wmma_f32_16x16x32_bf16(
            /*neg_a=*/false, afrag[mt], /*neg_b=*/false, bfrag[nt],
            /*c_mod=*/(short)0, acc[nt][mt],
            /*reuse_a=*/false, /*reuse_b=*/false);
      }
    }

    // own async writes done, then block-wide sync before switching buffers
    if (ks + 1 < kIters) wait_asynccnt0();
    __syncthreads();
  }

  // epilogue — C/D layout: VGPR v, lanes0-15 -> M=v, lanes16-31 -> M=v+8; N=lane&15
  const int rowHi = (lane < 16) ? 0 : 8;
  const int col   = lane & 15;
  #pragma unroll
  for (int nt = 0; nt < 2; ++nt) {
    const long gn = blockN + wc * 32 + nt * 16 + col;
    float bv = 0.0f;
    if (F32OUT) bv = bias[gn];               // one bias load per nt-tile
    #pragma unroll
    for (int mt = 0; mt < 4; ++mt) {
      const long gm = blockM + wr * 64 + mt * 16 + rowHi;
      v8f a = acc[nt][mt];
      #pragma unroll
      for (int v = 0; v < 8; ++v) {
        const long m = gm + v;
        if (F32OUT) outF[m * ldo + gn] = a[v] + bv;
        else        outB[m * ldo + gn] = (__bf16)a[v];
      }
    }
  }
}

// ---------------- launch ----------------
extern "C" void kernel_launch(void* const* d_in, const int* in_sizes, int n_in,
                              void* d_out, int out_size, void* d_ws, size_t ws_size,
                              hipStream_t stream) {
  const float* x    = (const float*)d_in[0];
  const float* lB   = (const float*)d_in[1];
  const float* lA   = (const float*)d_in[2];
  const float* sval = (const float*)d_in[3];
  const int*   sidx = (const int*)d_in[4];
  const float* bias = (const float*)d_in[5];
  const int nnz = in_sizes[3];

  // workspace layout (bf16): Xcat [N_TOK][KX], Wcat [OUT_F][KX], Abf [RANK][IN_F]
  char* ws = (char*)d_ws;
  const size_t XCAT_B = (size_t)N_TOK * KX * 2;   // 69,206,016
  const size_t WCAT_B = (size_t)OUT_F * KX * 2;   // 34,603,008
  __bf16* Xcat = (__bf16*)ws;
  __bf16* Wcat = (__bf16*)(ws + XCAT_B);
  __bf16* Abf  = (__bf16*)(ws + XCAT_B + WCAT_B);

  // 1) build bf16 operands
  {
    long tw = (long)OUT_F * (KX / 4);
    init_wcat<<<(unsigned)((tw + 255) / 256), 256, 0, stream>>>(lB, Wcat);
    long ta = (long)RANK * (IN_F / 4);
    init_abf<<<(unsigned)((ta + 255) / 256), 256, 0, stream>>>(lA, Abf);
    long tx = (long)N_TOK * (IN_F / 4);
    init_xcat<<<(unsigned)((tx + 255) / 256), 256, 0, stream>>>(x, Xcat);
  }

  // 2) scatter sorted COO into Wcat (no atomics; run-leader sums duplicates)
  scatter_sorted<<<(nnz + 255) / 256, 256, 0, stream>>>(sval, sidx, Wcat, nnz);

  // 3) LoRA first GEMM: t = x_bf16 @ (A/128)^T  -> Xcat columns [4096..4223]
  gemm_wmma_bf16<false><<<dim3(N_TOK / 128, RANK / 128), 256, 0, stream>>>(
      Xcat, KX, Abf, IN_F, IN_F / 32,
      nullptr, Xcat + IN_F, KX, nullptr);

  // 4) main fused GEMM: out = Xcat @ Wcat^T + bias  (K = 4224)
  gemm_wmma_bf16<true><<<dim3(N_TOK / 128, OUT_F / 128), 256, 0, stream>>>(
      Xcat, KX, Wcat, KX, KX / 32,
      (float*)d_out, nullptr, OUT_F, bias);
}
